// GATT_78142634983464
// MI455X (gfx1250) — matrix operand order repaired and checked by reading, not compiled, not run on hardware
//
#include <hip/hip_runtime.h>

// MI455X / gfx1250, wave32. All matmuls via V_WMMA_F32_16X16X32_BF16.
// Softmax row-max via DPP16 VALU reduction; row-sum via WMMA against a
// ones-matrix (keeps the hot loop on the matrix pipe, no ds_bpermute chains).

#define DEVINL __device__ __forceinline__

typedef __attribute__((ext_vector_type(16))) __bf16 v16bf;
typedef __attribute__((ext_vector_type(8)))  __bf16 v8bf;
typedef __attribute__((ext_vector_type(8)))  float  v8f;

constexpr int Cc  = 256;     // channels
constexpr int Hn  = 4;       // heads
constexpr int HID = 64;      // per-head dim
constexpr int Sn  = 2048;    // sequence
constexpr int Bn  = 8;       // batch
constexpr int Rn  = Bn * Sn; // 16384 rows

DEVINL __bf16 f2bf(float f) {
  union { float f; unsigned u; } a; a.f = f;
  unsigned u = a.u;
  unsigned r = (u + 0x7FFFu + ((u >> 16) & 1u)) >> 16;   // RNE
  union { unsigned short s; __bf16 b; } o; o.s = (unsigned short)r;
  return o.b;
}

DEVINL float gelu_f(float t) {
  return 0.5f * t * (1.0f + erff(t * 0.70710678118654752f));
}

DEVINL v8f wmma_bf16(v16bf a, v16bf b, v8f c) {
  return __builtin_amdgcn_wmma_f32_16x16x32_bf16(
      false, a, false, b, (short)0, c, false, false);
}

// 16-lane-group max reduction in pure VALU DPP16:
// quad xor1, quad xor2, row_half_mirror (xor7), row_mirror (xor15).
DEVINL float max_dpp16(float x) {
  union fi { float f; int i; };
  fi a; a.f = x;
  fi t;
  t.i = __builtin_amdgcn_update_dpp(0, a.i, 0xB1,  0xF, 0xF, true);  // quad_perm(1,0,3,2)
  a.f = fmaxf(a.f, t.f);
  t.i = __builtin_amdgcn_update_dpp(0, a.i, 0x4E,  0xF, 0xF, true);  // quad_perm(2,3,0,1)
  a.f = fmaxf(a.f, t.f);
  t.i = __builtin_amdgcn_update_dpp(0, a.i, 0x141, 0xF, 0xF, true);  // row_half_mirror
  a.f = fmaxf(a.f, t.f);
  t.i = __builtin_amdgcn_update_dpp(0, a.i, 0x140, 0xF, 0xF, true);  // row_mirror
  a.f = fmaxf(a.f, t.f);
  return a.f;
}

// ---- fragment loaders (CDNA5 ISA §7.12.2); all contiguous/vector ----

// A (16x32) row-major source: per-lane data = row[hi*16 .. +15] (contiguous K)
DEVINL v16bf load_afrag(const __bf16* __restrict__ p, int ld) {
  int lane = threadIdx.x & 31;
  const __bf16* row = p + (size_t)(lane & 15) * ld + ((lane >> 4) & 1) * 16;
  v8bf lo = *(const v8bf*)row;
  v8bf hi = *(const v8bf*)(row + 8);
  return __builtin_shufflevector(lo, hi, 0, 1, 2, 3, 4, 5, 6, 7,
                                 8, 9, 10, 11, 12, 13, 14, 15);
}

// B (32x16), element (k,n) at p[n*ld + k]: per-lane data = 16 contiguous bf16
// (lanes 0-15 hold K 0-15, lanes 16-31 hold K 16-31)
DEVINL v16bf load_bfrag_kc(const __bf16* __restrict__ p, int ld) {
  int lane = threadIdx.x & 31;
  return *(const v16bf*)(p + (size_t)(lane & 15) * ld + ((lane >> 4) & 1) * 16);
}

// ---------------- elementwise / LN kernels ----------------

__global__ void copy_f32(const float* __restrict__ in, float* __restrict__ out, int n) {
  int i = blockIdx.x * blockDim.x + threadIdx.x;
  if (i < n) out[i] = in[i];
}

// MODE 0: y=LN(in); xio=y, obf=bf16(y)                (outer pre-norm)
// MODE 1: y=gelu(LN(in)); obf=bf16(y)                 (ODE mid)
// MODE 2: y=gelu(LN(in)); xio+=y; obf=bf16(xio)       (GAT residual)
// MODE 3: y=gelu(LN(in)); xio+=y                      (ODE residual)
template <int MODE>
__global__ void ln_row(const float* __restrict__ in, const float* __restrict__ gw,
                       const float* __restrict__ bw, float* __restrict__ xio,
                       __bf16* __restrict__ obf) {
  int row  = (int)((blockIdx.x * blockDim.x + threadIdx.x) >> 5);
  int lane = threadIdx.x & 31;
  if (row >= Rn) return;
  const float* r = in + (size_t)row * Cc;
  float v[8], s = 0.f, sq = 0.f;
#pragma unroll
  for (int i = 0; i < 8; ++i) {
    v[i] = r[lane * 8 + i];
    s += v[i];
    sq += v[i] * v[i];
  }
#pragma unroll
  for (int off = 16; off >= 1; off >>= 1) {
    s  += __shfl_xor(s, off, 32);
    sq += __shfl_xor(sq, off, 32);
  }
  float mean = s * (1.f / Cc);
  float inv  = rsqrtf(sq * (1.f / Cc) - mean * mean + 1e-5f);
#pragma unroll
  for (int i = 0; i < 8; ++i) {
    int c = lane * 8 + i;
    float y = (v[i] - mean) * inv * gw[c] + bw[c];
    size_t idx = (size_t)row * Cc + c;
    if (MODE == 0) {
      xio[idx] = y;
      obf[idx] = f2bf(y);
    } else {
      float t = gelu_f(y);
      if (MODE == 1) {
        obf[idx] = f2bf(t);
      } else {
        float nx = xio[idx] + t;
        xio[idx] = nx;
        if (MODE == 2) obf[idx] = f2bf(nx);
      }
    }
  }
}

// ---------------- weight packing (fp32 -> bf16, fragment-major swizzle) ----------------
// Output layout: frag[(kt*16 + nt)*32 + lane][16 elems], where the B-fragment
// element (k,n) is k = kt*32 + (lane>>4)*16 + e, n = nt*16 + (lane&15).
// proj=1: source W is (H, C, HID) and n indexes the concatenated heads.
__global__ void pack_w_frag(const float* __restrict__ W, __bf16* __restrict__ Wp, int proj) {
  int i = blockIdx.x * blockDim.x + threadIdx.x;  // over 256*256
  if (i >= Cc * Cc) return;
  int e    = i & 15;
  int lane = (i >> 4) & 31;
  int nt   = (i >> 9) & 15;
  int kt   = i >> 13;
  int k = kt * 32 + ((lane >> 4) & 1) * 16 + e;
  int n = nt * 16 + (lane & 15);
  float src = proj ? W[(((size_t)(n >> 6)) * Cc + k) * HID + (n & 63)]
                   : W[(size_t)k * Cc + n];
  Wp[i] = f2bf(src);
}

// ---------------- GEMM: [16384 x 256] x [256 x 256] + bias ----------------
// A: bf16 [M][256] row-major. Bm: fragment-major packed weights (see pack_w_frag).
// MODE 0: fp32 out0 [M][256]
// MODE 1: bf16 out0 = h [B][H][S][HID], bf16 out1 = h^T [B][H][HID][S]
template <int MODE>
__global__ void gemm256(const __bf16* __restrict__ A, const __bf16* __restrict__ Bm,
                        const float* __restrict__ bias, void* __restrict__ out0,
                        void* __restrict__ out1) {
  int wave = threadIdx.x >> 5;
  int lane = threadIdx.x & 31;
  int m0 = (blockIdx.x * 8 + wave) * 16;
  int n0 = blockIdx.y * 64;
  int nt0 = n0 >> 4;

  v8f acc[4] = {};
  const __bf16* arow = A + (size_t)m0 * Cc;
  const v16bf* Bf = (const v16bf*)Bm;
#pragma unroll
  for (int kt = 0; kt < 8; ++kt) {
    if (kt < 7) __builtin_prefetch(arow + kt * 32 + 32, 0, 3);  // global_prefetch
    v16bf a = load_afrag(arow + kt * 32, Cc);
#pragma unroll
    for (int j = 0; j < 4; ++j) {
      v16bf b = Bf[(size_t)(kt * 16 + nt0 + j) * 32 + lane];
      acc[j] = wmma_bf16(a, b, acc[j]);
    }
  }

  int hi = (lane >> 4) & 1, n = lane & 15;
#pragma unroll
  for (int j = 0; j < 4; ++j) {
    float bv = bias[n0 + j * 16 + n];
#pragma unroll
    for (int i = 0; i < 8; ++i) {
      int m = m0 + hi * 8 + i;
      int nn = n0 + j * 16 + n;
      float val = acc[j][i] + bv;
      if (MODE == 0) {
        ((float*)out0)[(size_t)m * Cc + nn] = val;
      } else {
        int b_ = m >> 11, s = m & (Sn - 1);
        int h_ = nn >> 6, d = nn & 63;
        __bf16 bfv = f2bf(val);
        size_t bh = (size_t)(b_ * Hn + h_);
        ((__bf16*)out0)[(bh * Sn + s) * HID + d] = bfv;
        ((__bf16*)out1)[(bh * HID + d) * Sn + s] = bfv;  // transposed copy for PV
      }
    }
  }
}

// ---------------- flash attention: scores=h h^T, softmax, o=attn h ----------------
// h: bf16 [B][H][S][64]; ht: bf16 [B][H][64][S]; o: fp32 [B][S][256] (concat heads)
// One wave per 16-query tile; 32 keys/step: 4 score WMMA + 4 PV WMMA + 1 sum WMMA.
__global__ void attn(const __bf16* __restrict__ h, const __bf16* __restrict__ ht,
                     float* __restrict__ o) {
  __shared__ __bf16 lds[4][16 * 32];  // per-wave P staging (D-layout -> A-layout)
  int gw   = blockIdx.x * 4 + (threadIdx.x >> 5);
  int lane = threadIdx.x & 31;
  int qtile = gw & (Sn / 16 - 1);
  int bh    = gw >> 7;           // 0..31
  int b_    = bh >> 2, head = bh & 3;
  const __bf16* hb  = h  + (size_t)bh * Sn * HID;
  const __bf16* htb = ht + (size_t)bh * HID * Sn;
  int q0 = qtile * 16;
  __bf16* pl = lds[threadIdx.x >> 5];

  // Q fragments (16 x 64), resident for the whole loop
  v16bf a0 = load_afrag(hb + (size_t)q0 * HID, HID);
  v16bf a1 = load_afrag(hb + (size_t)q0 * HID + 32, HID);

  // all-ones B fragment: rowsum(P) = P @ ones on the matrix pipe
  v16bf ones;
  {
    __bf16 one = f2bf(1.0f);
#pragma unroll
    for (int j = 0; j < 16; ++j) ones[j] = one;
  }

  int hi = (lane >> 4) & 1, nl = lane & 15;
  float st_m[8];
  v8f lsum = {};
  v8f acc[4] = {};
#pragma unroll
  for (int i = 0; i < 8; ++i) st_m[i] = -INFINITY;

  for (int key0 = 0; key0 < Sn; key0 += 32) {
    // scores S = Q K^T for 32 keys -> two 16x16 fp32 fragments
    v8f s0 = {}, s1 = {};
    {
      v16bf bk;
      bk = load_bfrag_kc(hb + (size_t)key0 * HID, HID);              s0 = wmma_bf16(a0, bk, s0);
      bk = load_bfrag_kc(hb + (size_t)key0 * HID + 32, HID);         s0 = wmma_bf16(a1, bk, s0);
      bk = load_bfrag_kc(hb + (size_t)(key0 + 16) * HID, HID);       s1 = wmma_bf16(a0, bk, s1);
      bk = load_bfrag_kc(hb + (size_t)(key0 + 16) * HID + 32, HID);  s1 = wmma_bf16(a1, bk, s1);
    }

    // online softmax: row max via pure-VALU DPP16 reduction (16-lane groups)
#pragma unroll
    for (int i = 0; i < 8; ++i) {
      float nm = max_dpp16(fmaxf(s0[i], s1[i]));
      float mnew = fmaxf(st_m[i], nm);
      float sc = __expf(st_m[i] - mnew);
      st_m[i] = mnew;
      lsum[i] *= sc;
#pragma unroll
      for (int j = 0; j < 4; ++j) acc[j][i] *= sc;
    }

    // P = exp(S - m), staged through per-wave LDS (same-wave DS: in-order)
#pragma unroll
    for (int i = 0; i < 8; ++i) {
      float p0 = __expf(s0[i] - st_m[i]);
      float p1 = __expf(s1[i] - st_m[i]);
      int r = i + hi * 8;
      pl[r * 32 + nl]      = f2bf(p0);
      pl[r * 32 + 16 + nl] = f2bf(p1);
    }

    v16bf pa = load_afrag(pl, 32);
    lsum = wmma_bf16(pa, ones, lsum);  // row sums accumulate on the matrix pipe

    // o += P(16x32) @ V(32x64); V fragments contiguous via transposed copy
#pragma unroll
    for (int j = 0; j < 4; ++j) {
      v16bf vb = load_bfrag_kc(htb + (size_t)(j * 16) * Sn + key0, Sn);
      acc[j] = wmma_bf16(pa, vb, acc[j]);
    }
  }

  // normalize and store concat-head output
#pragma unroll
  for (int i = 0; i < 8; ++i) {
    float inv = 1.f / lsum[i];
    int m = q0 + hi * 8 + i;
#pragma unroll
    for (int j = 0; j < 4; ++j)
      o[((size_t)b_ * Sn + m) * Cc + head * HID + j * 16 + nl] = acc[j][i] * inv;
  }
}

// ---------------- host orchestration ----------------

extern "C" void kernel_launch(void* const* d_in, const int* in_sizes, int n_in,
                              void* d_out, int out_size, void* d_ws, size_t ws_size,
                              hipStream_t stream) {
  (void)in_sizes; (void)n_in; (void)out_size; (void)ws_size;
  const float* x0 = (const float*)d_in[0];
  float* x = (float*)d_out;  // running activation, fp32 [B][S][C]

  char* ws = (char*)d_ws;
  __bf16* xbf  = (__bf16*)(ws);                               // 8 MB: bf16 activations (A operand)
  __bf16* hbf  = (__bf16*)(ws + (size_t)Rn * Cc * 2);         // 8 MB: h (bhsd) / ODE mid bf16
  __bf16* htbf = (__bf16*)(ws + (size_t)Rn * Cc * 4);         // 8 MB: h^T (bhds)
  float*  tf   = (float*)(ws + (size_t)Rn * Cc * 6);          // 16 MB: attn out / ODE GEMM out
  __bf16* Wp   = (__bf16*)(ws + (size_t)Rn * Cc * 10);        // 128 KB: packed bf16 weights

  const int NE = Rn * Cc;  // 4,194,304
  copy_f32<<<NE / 256, 256, 0, stream>>>(x0, x, NE);

  dim3 lnGrid(Rn / 8), lnBlk(256);          // 8 rows per 256-thread block
  dim3 gGrid(Rn / 128, Cc / 64), gBlk(256); // wave = 16x64 tile
  dim3 aGrid(Bn * Hn * (Sn / 16) / 4), aBlk(128);
  dim3 pGrid(Cc * Cc / 256), pBlk(256);

  for (int p = 0; p < 3; ++p) {
    int base = 1 + p * 14;
    const float* norm_g = (const float*)d_in[base + 0];
    const float* norm_b = (const float*)d_in[base + 1];
    const float* Wgat   = (const float*)d_in[base + 2];
    const float* bWgat  = (const float*)d_in[base + 3];
    const float* gat_g  = (const float*)d_in[base + 4];
    const float* gat_b  = (const float*)d_in[base + 5];

    // x = LN(x); xbf = bf16(x)
    ln_row<0><<<lnGrid, lnBlk, 0, stream>>>(x, norm_g, norm_b, x, xbf);

    // h = x @ Wcat + bW  (bf16 WMMA GEMM, writes h and h^T)
    pack_w_frag<<<pGrid, pBlk, 0, stream>>>(Wgat, Wp, 1);
    gemm256<1><<<gGrid, gBlk, 0, stream>>>(xbf, Wp, bWgat, (void*)hbf, (void*)htbf);

    // o = softmax(h h^T) h, heads concatenated
    attn<<<aGrid, aBlk, 0, stream>>>(hbf, htbf, tf);

    // x += gelu(LN(o)); xbf = bf16(x)
    ln_row<2><<<lnGrid, lnBlk, 0, stream>>>(tf, gat_g, gat_b, x, xbf);

    // ODE block 0: t = x @ W0 + b0; u = gelu(LN(t)) -> hbf (bf16)
    pack_w_frag<<<pGrid, pBlk, 0, stream>>>((const float*)d_in[base + 6], Wp, 0);
    gemm256<0><<<gGrid, gBlk, 0, stream>>>(xbf, Wp, (const float*)d_in[base + 7],
                                           (void*)tf, (void*)tf);
    ln_row<1><<<lnGrid, lnBlk, 0, stream>>>(tf, (const float*)d_in[base + 8],
                                            (const float*)d_in[base + 9], nullptr, hbf);

    // ODE block 1: t = u @ W1 + b1; x += gelu(LN(t))
    pack_w_frag<<<pGrid, pBlk, 0, stream>>>((const float*)d_in[base + 10], Wp, 0);
    gemm256<0><<<gGrid, gBlk, 0, stream>>>(hbf, Wp, (const float*)d_in[base + 11],
                                           (void*)tf, (void*)tf);
    ln_row<3><<<lnGrid, lnBlk, 0, stream>>>(tf, (const float*)d_in[base + 12],
                                            (const float*)d_in[base + 13], x, nullptr);
  }
}
